// QLSTMGen_65481071396633
// MI455X (gfx1250) — matrix operand order repaired and checked by reading, not compiled, not run on hardware
//
#include <hip/hip_runtime.h>
#include <hip/hip_bf16.h>
#include <math.h>

// QLSTM on MI455X (gfx1250).
// Key algorithmic restructure for this chip: the x-half of K in comb@W.T is
// h-independent, so it is hoisted into ONE fully-parallel WMMA GEMM
// ([131072,256]x[256,1024], 68.7 GFLOP) that can saturate the matrix pipes,
// stored in WMMA C-tile layout. The serial 512-step recurrence then only does
// the h-half (K=256) per step -- half the critical path -- with LDS-staged f16
// h-panels and double-buffered L2-resident f16 weights.
// Fallback (ws too small for 512MB xz): full-K per-step kernel.

typedef __attribute__((ext_vector_type(16))) _Float16 v16h;
typedef __attribute__((ext_vector_type(8)))  _Float16 v8h;
typedef __attribute__((ext_vector_type(8)))  float    v8f;

#define B_     256
#define HID_   256
#define IN_    256
#define D_     512   // IN_ + HID_
#define SEQ_   512
#define WELEM  131072  // 256*512 per gate
#define LDSPAD 8       // halves; keeps 16B alignment, staggers banks

__device__ __forceinline__ float sigmoidf_(float x) {
  return 1.0f / (1.0f + __expf(-x));
}
__device__ __forceinline__ float qgate_(float z, float th, float sc) {
  // simulated quantum gate: sigmoid(sin(z*theta)*scale)
  return sigmoidf_(__sinf(z * th) * sc);
}

__device__ __forceinline__ v16h packA(const _Float16* ap) {
  // A 16x32 in this code's contiguous k order (B matches): lane<16 holds K {0..15}; lane>=16 holds {16..31}
  const v8h a0 = *(const v8h*)ap;          // ds_load_b128
  const v8h a1 = *(const v8h*)(ap + 8);    // ds_load_b128
  v16h a;
  #pragma unroll
  for (int i = 0; i < 8; ++i) { a[i] = a0[i]; a[8 + i] = a1[i]; }
  return a;
}

__global__ __launch_bounds__(256) void qlstm_cvt(const float* __restrict__ s,
                                                 _Float16* __restrict__ d, int n) {
  int i = blockIdx.x * blockDim.x + threadIdx.x;
  if (i < n) d[i] = (_Float16)s[i];
}

// ---------------------------------------------------------------------------
// One-shot x-projection: xz[mt,gctile] = x_rows(mt) @ Wx(gctile).T
// Output in WMMA C-tile layout: xz[(tile*32 + lane)*8 + v], tile = mt*64+gctile.
// Grid: (SEQ*B/16) * 16 = 131072 blocks x 128 threads (4 waves, 4 col-tiles).
// ---------------------------------------------------------------------------
__global__ __launch_bounds__(128) void qlstm_xproj(
    const float*    __restrict__ xall,   // [SEQ*B, IN] f32
    const _Float16* __restrict__ w16,    // [4][HID][D] f16
    float*          __restrict__ xz)     // tile layout, f32
{
  __shared__ _Float16 As[16][IN_ + LDSPAD];
  const int tid  = threadIdx.x;
  const int lane = tid & 31;
  const int wave = tid >> 5;       // 0..3
  const int mt   = blockIdx.x >> 4;    // 0..8191 row tile over SEQ*B
  const int ng   = blockIdx.x & 15;    // col group of 64
  const int l15  = lane & 15;
  const int lh   = lane >> 4;

  { // stage 16 x-rows (f32 -> f16) into LDS
    const int row  = tid >> 3;
    const int cs   = (tid & 7) << 5;
    const size_t grow = (size_t)mt * 16 + row;
    const float4* xp = (const float4*)(xall + grow * IN_ + cs);
    alignas(16) _Float16 tmp[32];
    #pragma unroll
    for (int q = 0; q < 8; ++q) {
      float4 v = xp[q];
      tmp[4*q+0] = (_Float16)v.x; tmp[4*q+1] = (_Float16)v.y;
      tmp[4*q+2] = (_Float16)v.z; tmp[4*q+3] = (_Float16)v.w;
    }
    #pragma unroll
    for (int q = 0; q < 4; ++q)
      *(v8h*)&As[row][cs + 8*q] = *(const v8h*)&tmp[8*q];
  }
  __syncthreads();

  const int gctile = ng * 4 + wave;    // 0..63
  const int gc     = gctile * 16 + l15;  // 0..1023 over 4*HID
  const int gate   = gc >> 8;
  const int jcol   = gc & 255;
  const size_t wbase = (size_t)gate * WELEM + (size_t)jcol * D_ + (size_t)(lh * 16);

  v8f acc = {};
  v16h bnxt = *(const v16h*)(w16 + wbase);   // x-part of K: offsets 0..255
  #pragma unroll
  for (int k = 0; k < IN_; k += 32) {
    const v16h bc = bnxt;
    if (k + 32 < IN_) bnxt = *(const v16h*)(w16 + wbase + k + 32);
    const v16h a = packA(&As[l15][k + lh * 16]);
    acc = __builtin_amdgcn_wmma_f32_16x16x32_f16(false, a, false, bc, (short)0, acc, false, false);
  }
  *(v8f*)(xz + ((size_t)(mt * 64 + gctile) * 32 + lane) * 8) = acc;   // b128 x2
}

// ---------------------------------------------------------------------------
// Fast per-step kernel: K=256 (h-half only), accumulators seeded from xz tiles.
// Grid: 64 blocks x 128 threads.
// ---------------------------------------------------------------------------
__global__ __launch_bounds__(128) void qlstm_step_fast(
    const float*    __restrict__ xz_t,     // this step's tiles (16*64 tiles)
    const float*    __restrict__ xz_next,  // next step's tiles (prefetch only)
    const _Float16* __restrict__ hprev16,  // [B, HID] f16
    _Float16*       __restrict__ hnext16,  // [B, HID] f16
    float*          __restrict__ cbuf,     // [B, HID] f32 (tile-private RMW)
    const _Float16* __restrict__ w16,      // [4][HID][D] f16
    const float* __restrict__ bf_, const float* __restrict__ thf_, const float* __restrict__ scf_,
    const float* __restrict__ bi_, const float* __restrict__ thi_, const float* __restrict__ sci_,
    const float* __restrict__ bu_, const float* __restrict__ thu_, const float* __restrict__ scu_,
    const float* __restrict__ bo_, const float* __restrict__ tho_, const float* __restrict__ sco_,
    float* __restrict__ outt)
{
  __shared__ _Float16 As[16][HID_ + LDSPAD];   // h-panel only

  const int tid     = threadIdx.x;
  const int lane    = tid & 31;
  const int wave    = tid >> 5;
  const int rowTile = blockIdx.x >> 2;                 // 0..15
  const int rowBase = rowTile << 4;
  const int colTile = ((blockIdx.x & 3) << 2) + wave;  // 0..15
  const int col     = (colTile << 4) + (lane & 15);
  const int l15     = lane & 15;
  const int lh      = lane >> 4;

  { // stage 16 h-rows (f16) into LDS; prefetch next step's xz region
    const int row = tid >> 3;
    const int cs  = (tid & 7) << 5;
    const _Float16* hp = hprev16 + (size_t)(rowBase + row) * HID_ + cs;
    #pragma unroll
    for (int q = 0; q < 4; ++q)
      *(v8h*)&As[row][cs + 8*q] = *(const v8h*)(hp + 8*q);
    __builtin_prefetch(xz_next + ((size_t)(rowTile * 64 + colTile) * 32 + lane) * 8, 0, 1);
  }
  __syncthreads();

  // Seed accumulators with the precomputed x-projection (WMMA C input).
  const size_t t0 = ((size_t)(rowTile * 64 + colTile) * 32 + lane) * 8;
  v8f accF = *(const v8f*)(xz_t + t0);
  v8f accI = *(const v8f*)(xz_t + t0 + (size_t)16 * 32 * 8);
  v8f accU = *(const v8f*)(xz_t + t0 + (size_t)32 * 32 * 8);
  v8f accO = *(const v8f*)(xz_t + t0 + (size_t)48 * 32 * 8);

  const _Float16* wF = w16;
  const _Float16* wI = w16 + 1 * WELEM;
  const _Float16* wU = w16 + 2 * WELEM;
  const _Float16* wO = w16 + 3 * WELEM;
  const size_t wbase = (size_t)col * D_ + IN_ + (size_t)(lh * 16);  // h-part of K

  v16h bF = *(const v16h*)(wF + wbase);
  v16h bI = *(const v16h*)(wI + wbase);
  v16h bU = *(const v16h*)(wU + wbase);
  v16h bO = *(const v16h*)(wO + wbase);

  #pragma unroll 2
  for (int k = 0; k < HID_; k += 32) {
    const v16h cF = bF, cI = bI, cU = bU, cO = bO;
    if (k + 32 < HID_) {
      const size_t nofs = wbase + (size_t)(k + 32);
      bF = *(const v16h*)(wF + nofs);
      bI = *(const v16h*)(wI + nofs);
      bU = *(const v16h*)(wU + nofs);
      bO = *(const v16h*)(wO + nofs);
    }
    const v16h a = packA(&As[l15][k + lh * 16]);
    accF = __builtin_amdgcn_wmma_f32_16x16x32_f16(false, a, false, cF, (short)0, accF, false, false);
    accI = __builtin_amdgcn_wmma_f32_16x16x32_f16(false, a, false, cI, (short)0, accI, false, false);
    accU = __builtin_amdgcn_wmma_f32_16x16x32_f16(false, a, false, cU, (short)0, accU, false, false);
    accO = __builtin_amdgcn_wmma_f32_16x16x32_f16(false, a, false, cO, (short)0, accO, false, false);
  }

  const float bF0 = bf_[col], tF0 = thf_[col], sF0 = scf_[col];
  const float bI0 = bi_[col], tI0 = thi_[col], sI0 = sci_[col];
  const float bU0 = bu_[col], tU0 = thu_[col], sU0 = scu_[col];
  const float bO0 = bo_[col], tO0 = tho_[col], sO0 = sco_[col];

  #pragma unroll
  for (int v = 0; v < 8; ++v) {
    const int row = rowBase + v + lh * 8;
    const size_t idx = (size_t)row * HID_ + col;
    const float f = sigmoidf_(qgate_(accF[v] + bF0, tF0, sF0));
    const float i = sigmoidf_(qgate_(accI[v] + bI0, tI0, sI0));
    const float g = tanhf(qgate_(accU[v] + bU0, tU0, sU0));
    const float o = sigmoidf_(qgate_(accO[v] + bO0, tO0, sO0));
    const float cn = f * cbuf[idx] + i * g;
    const float hn = o * tanhf(cn);
    cbuf[idx]    = cn;
    hnext16[idx] = (_Float16)hn;
    outt[idx]    = hn;
  }
}

// ---------------------------------------------------------------------------
// Fallback per-step kernel (ws too small for xz): full K=512 with x staged.
// ---------------------------------------------------------------------------
__global__ __launch_bounds__(128) void qlstm_step_full(
    const float*    __restrict__ x,
    const float*    __restrict__ xnext,
    const _Float16* __restrict__ hprev16,
    _Float16*       __restrict__ hnext16,
    float*          __restrict__ cbuf,
    const _Float16* __restrict__ w16,
    const float* __restrict__ bf_, const float* __restrict__ thf_, const float* __restrict__ scf_,
    const float* __restrict__ bi_, const float* __restrict__ thi_, const float* __restrict__ sci_,
    const float* __restrict__ bu_, const float* __restrict__ thu_, const float* __restrict__ scu_,
    const float* __restrict__ bo_, const float* __restrict__ tho_, const float* __restrict__ sco_,
    float* __restrict__ outt)
{
  __shared__ _Float16 As[16][D_ + LDSPAD];

  const int tid     = threadIdx.x;
  const int lane    = tid & 31;
  const int wave    = tid >> 5;
  const int rowBase = (blockIdx.x >> 2) << 4;
  const int col     = ((blockIdx.x & 3) << 6) + (wave << 4) + (lane & 15);
  const int l15     = lane & 15;
  const int lh      = lane >> 4;

  {
    const int row  = tid >> 3;
    const int cs   = (tid & 7) << 5;
    const int grow = rowBase + row;
    const float4* xp = (const float4*)(x + (size_t)grow * IN_ + cs);
    alignas(16) _Float16 tmp[32];
    #pragma unroll
    for (int q = 0; q < 8; ++q) {
      float4 v = xp[q];
      tmp[4*q+0] = (_Float16)v.x; tmp[4*q+1] = (_Float16)v.y;
      tmp[4*q+2] = (_Float16)v.z; tmp[4*q+3] = (_Float16)v.w;
    }
    #pragma unroll
    for (int q = 0; q < 4; ++q)
      *(v8h*)&As[row][cs + 8*q] = *(const v8h*)&tmp[8*q];
    const _Float16* hp = hprev16 + (size_t)grow * HID_ + cs;
    #pragma unroll
    for (int q = 0; q < 4; ++q)
      *(v8h*)&As[row][IN_ + cs + 8*q] = *(const v8h*)(hp + 8*q);
    __builtin_prefetch(xnext + (size_t)grow * IN_ + cs, 0, 1);
  }
  __syncthreads();

  v8f accF = {}, accI = {}, accU = {}, accO = {};
  const _Float16* wF = w16;
  const _Float16* wI = w16 + 1 * WELEM;
  const _Float16* wU = w16 + 2 * WELEM;
  const _Float16* wO = w16 + 3 * WELEM;
  const size_t wbase = (size_t)col * D_ + (size_t)(lh * 16);

  v16h bF = *(const v16h*)(wF + wbase);
  v16h bI = *(const v16h*)(wI + wbase);
  v16h bU = *(const v16h*)(wU + wbase);
  v16h bO = *(const v16h*)(wO + wbase);

  #pragma unroll 2
  for (int k = 0; k < D_; k += 32) {
    const v16h cF = bF, cI = bI, cU = bU, cO = bO;
    if (k + 32 < D_) {
      const size_t nofs = wbase + (size_t)(k + 32);
      bF = *(const v16h*)(wF + nofs);
      bI = *(const v16h*)(wI + nofs);
      bU = *(const v16h*)(wU + nofs);
      bO = *(const v16h*)(wO + nofs);
    }
    const v16h a = packA(&As[l15][k + lh * 16]);
    accF = __builtin_amdgcn_wmma_f32_16x16x32_f16(false, a, false, cF, (short)0, accF, false, false);
    accI = __builtin_amdgcn_wmma_f32_16x16x32_f16(false, a, false, cI, (short)0, accI, false, false);
    accU = __builtin_amdgcn_wmma_f32_16x16x32_f16(false, a, false, cU, (short)0, accU, false, false);
    accO = __builtin_amdgcn_wmma_f32_16x16x32_f16(false, a, false, cO, (short)0, accO, false, false);
  }

  const float bF0 = bf_[col], tF0 = thf_[col], sF0 = scf_[col];
  const float bI0 = bi_[col], tI0 = thi_[col], sI0 = sci_[col];
  const float bU0 = bu_[col], tU0 = thu_[col], sU0 = scu_[col];
  const float bO0 = bo_[col], tO0 = tho_[col], sO0 = sco_[col];

  #pragma unroll
  for (int v = 0; v < 8; ++v) {
    const int row = rowBase + v + lh * 8;
    const size_t idx = (size_t)row * HID_ + col;
    const float f = sigmoidf_(qgate_(accF[v] + bF0, tF0, sF0));
    const float i = sigmoidf_(qgate_(accI[v] + bI0, tI0, sI0));
    const float g = tanhf(qgate_(accU[v] + bU0, tU0, sU0));
    const float o = sigmoidf_(qgate_(accO[v] + bO0, tO0, sO0));
    const float cn = f * cbuf[idx] + i * g;
    const float hn = o * tanhf(cn);
    cbuf[idx]    = cn;
    hnext16[idx] = (_Float16)hn;
    outt[idx]    = hn;
  }
}

extern "C" void kernel_launch(void* const* d_in, const int* in_sizes, int n_in,
                              void* d_out, int out_size, void* d_ws, size_t ws_size,
                              hipStream_t stream) {
  (void)in_sizes; (void)n_in; (void)out_size;
  const float* xs  = (const float*)d_in[0];
  const float* h0  = (const float*)d_in[1];
  const float* c0  = (const float*)d_in[2];
  const float* Wf  = (const float*)d_in[3];
  const float* bf  = (const float*)d_in[4];
  const float* thf = (const float*)d_in[5];
  const float* scf = (const float*)d_in[6];
  const float* Wi  = (const float*)d_in[7];
  const float* bi  = (const float*)d_in[8];
  const float* thi = (const float*)d_in[9];
  const float* sci = (const float*)d_in[10];
  const float* Wu  = (const float*)d_in[11];
  const float* bu  = (const float*)d_in[12];
  const float* thu = (const float*)d_in[13];
  const float* scu = (const float*)d_in[14];
  const float* Wo  = (const float*)d_in[15];
  const float* bo  = (const float*)d_in[16];
  const float* tho = (const float*)d_in[17];
  const float* sco = (const float*)d_in[18];
  float* out = (float*)d_out;

  const size_t BH = (size_t)B_ * HID_;

  // Workspace: [4*WELEM f16 W][h16 ping][h16 pong][c f32][optional xz f32 512MB]
  char* ws = (char*)d_ws;
  _Float16* w16 = (_Float16*)ws;
  _Float16* hb0 = (_Float16*)(ws + (size_t)4 * WELEM * sizeof(_Float16));
  _Float16* hb1 = hb0 + BH;
  float*    cb  = (float*)(hb1 + BH);
  const size_t base_bytes = (size_t)4 * WELEM * 2 + 2 * BH * 2 + BH * 4;
  const size_t xz_bytes   = (size_t)SEQ_ * B_ * 4 * HID_ * sizeof(float);  // 512 MB
  const bool fast = ws_size >= base_bytes + xz_bytes;
  float* xz = fast ? (float*)(ws + base_bytes) : nullptr;
  const size_t xz_step = (size_t)16 * 64 * 32 * 8;  // floats per timestep (16 row-tiles)

  qlstm_cvt<<<WELEM / 256, 256, 0, stream>>>(Wf, w16 + 0 * WELEM, WELEM);
  qlstm_cvt<<<WELEM / 256, 256, 0, stream>>>(Wi, w16 + 1 * WELEM, WELEM);
  qlstm_cvt<<<WELEM / 256, 256, 0, stream>>>(Wu, w16 + 2 * WELEM, WELEM);
  qlstm_cvt<<<WELEM / 256, 256, 0, stream>>>(Wo, w16 + 3 * WELEM, WELEM);
  qlstm_cvt<<<(int)(BH / 256), 256, 0, stream>>>(h0, hb0, (int)BH);
  hipMemcpyAsync(cb, c0, BH * sizeof(float), hipMemcpyDeviceToDevice, stream);

  if (fast) {
    // One fully-parallel WMMA GEMM for the x-projection of ALL timesteps.
    qlstm_xproj<<<(SEQ_ * B_ / 16) * 16, 128, 0, stream>>>(xs, w16, xz);
  }

  for (int t = 0; t < SEQ_; ++t) {
    _Float16* hp = (t & 1) ? hb1 : hb0;
    _Float16* hn = (t & 1) ? hb0 : hb1;
    float* ot = out + (size_t)t * BH;
    if (fast) {
      const float* xzt = xz + (size_t)t * xz_step;
      const float* xzn = (t + 1 < SEQ_) ? xzt + xz_step : xzt;
      qlstm_step_fast<<<64, 128, 0, stream>>>(
          xzt, xzn, hp, hn, cb, w16,
          bf, thf, scf, bi, thi, sci, bu, thu, scu, bo, tho, sco, ot);
    } else {
      const float* xt = xs + (size_t)t * B_ * IN_;
      const float* xn = (t + 1 < SEQ_) ? xt + (size_t)B_ * IN_ : xt;
      qlstm_step_full<<<64, 128, 0, stream>>>(
          xt, xn, hp, hn, cb, w16,
          bf, thf, scf, bi, thi, sci, bu, thu, scu, bo, tho, sco, ot);
    }
  }

  // Final h (f32) is the last output slice; final c is in cb.
  hipMemcpyAsync(out + (size_t)SEQ_ * BH, out + (size_t)(SEQ_ - 1) * BH,
                 BH * sizeof(float), hipMemcpyDeviceToDevice, stream);
  hipMemcpyAsync(out + (size_t)SEQ_ * BH + BH, cb,
                 BH * sizeof(float), hipMemcpyDeviceToDevice, stream);
}
